// SwinTransformerBlock_283467842162
// MI455X (gfx1250) — matrix operand order repaired and checked by reading, not compiled, not run on hardware
//
#include <hip/hip_runtime.h>
#include <hip/hip_bf16.h>

// ---------------------------------------------------------------------------
// Swin Transformer block, MI455X (gfx1250, wave32, WMMA 16x16x32 f16,
// async global->LDS copies double-buffered under the WMMAs).
// B=16, H=W=56, C=256, NH=8, HD=32, WS=7, N=49, shift=3. M = 16*3136 = 50176.
// ---------------------------------------------------------------------------

typedef __attribute__((ext_vector_type(16))) _Float16 v16h;
typedef __attribute__((ext_vector_type(8)))  _Float16 v8h;
typedef __attribute__((ext_vector_type(8)))  float    v8f;
typedef __attribute__((ext_vector_type(4)))  int      v4i;

#define MROWS 50176
#define CDIM  256
#define SCALE_Q 0.17677669529663687f   // 32^-0.5

// ---- CDNA5 async global->LDS copy (ASYNCcnt) ------------------------------
// 16 bytes per lane, no VGPR staging. Builtin signature (probe-confirmed by
// the round-2 diagnostic): (int4 AS(1)*, int4 AS(3)*, imm offset, imm cpol).
typedef __attribute__((address_space(1))) v4i* glb_v4i_ptr;
typedef __attribute__((address_space(3))) v4i* lds_v4i_ptr;

static __device__ __forceinline__ void async_cp_b128(const _Float16* g, _Float16* l) {
#if __has_builtin(__builtin_amdgcn_global_load_async_to_lds_b128)
    __builtin_amdgcn_global_load_async_to_lds_b128(
        (glb_v4i_ptr)(uintptr_t)g,
        (lds_v4i_ptr)(uint32_t)(uintptr_t)l,
        0, 0);
#else
    *(v8h*)l = *(const v8h*)g;
#endif
}

template <int N>
static __device__ __forceinline__ void wait_async() {
#if __has_builtin(__builtin_amdgcn_s_wait_asynccnt)
    __builtin_amdgcn_s_wait_asynccnt(N);
#else
    asm volatile("s_wait_asynccnt %0" :: "n"(N) : "memory");
#endif
}

// ---- WMMA fragment loaders (contiguous K per lane half, same order in A and B) -----
// A (16x32, M x K): lane m=lane&15; elems 0..15 = K=2*kb..2*kb+15,
// kb = (lane>>4)*8.
static __device__ __forceinline__ v16h ld_afrag(const _Float16* row, int kb) {
    v16h a;
#pragma unroll
    for (int i = 0; i < 16; ++i) a[i] = row[2 * kb + i];
    return a;
}
// B (32x16, K x N) stored N-major (Bt[n][k]): lane n=lane&15 holds 16
// contiguous K at koff = (lane>>4)*16.
static __device__ __forceinline__ v16h ld_bfrag(const _Float16* p) {
    v16h b;
#pragma unroll
    for (int i = 0; i < 16; ++i) b[i] = p[i];
    return b;
}

static __device__ __forceinline__ float wave_sum(float v) {
#pragma unroll
    for (int m = 16; m > 0; m >>= 1) v += __shfl_xor(v, m, 32);
    return v;
}

// ---------------------------------------------------------------------------
// fp32 -> f16 weight conversion
// ---------------------------------------------------------------------------
__global__ __launch_bounds__(256) void cvt_f16_kernel(const float* __restrict__ in,
                                                      _Float16* __restrict__ out, int n) {
    int i = blockIdx.x * 256 + threadIdx.x;
    if (i < n) out[i] = (_Float16)in[i];
}

// ---------------------------------------------------------------------------
// LayerNorm. windowed=1: fuse cyclic shift(-3,-3) + 7x7 window partition into
// the f16 store (dest row = win*49 + pos). windowed=0: identity layout.
// One wave per 256-wide row; 8 rows per 256-thread block.
// ---------------------------------------------------------------------------
__global__ __launch_bounds__(256) void ln_kernel(const float* __restrict__ x,
                                                 const float* __restrict__ g,
                                                 const float* __restrict__ bb,
                                                 _Float16* __restrict__ out,
                                                 int windowed) {
    const int row  = blockIdx.x * 8 + (threadIdx.x >> 5);
    const int lane = threadIdx.x & 31;
    const float* xr = x + (size_t)row * CDIM + lane * 8;

    float v[8];
    float s = 0.f;
#pragma unroll
    for (int i = 0; i < 8; ++i) { v[i] = xr[i]; s += v[i]; }
    const float mean = wave_sum(s) * (1.0f / CDIM);
    float vs = 0.f;
#pragma unroll
    for (int i = 0; i < 8; ++i) { float d = v[i] - mean; vs += d * d; }
    const float rstd = rsqrtf(wave_sum(vs) * (1.0f / CDIM) + 1e-5f);

    size_t drow;
    if (windowed) {
        int b = row / 3136, hw = row % 3136, h = hw / 56, w = hw % 56;
        int hp = h - 3; if (hp < 0) hp += 56;           // roll(-3)
        int wp = w - 3; if (wp < 0) wp += 56;
        int win = b * 64 + (hp / 7) * 8 + (wp / 7);
        int nn  = (hp % 7) * 7 + (wp % 7);
        drow = (size_t)win * 49 + nn;
    } else {
        drow = (size_t)row;
    }
    _Float16* dst = out + drow * CDIM + lane * 8;
    const float* gp = g + lane * 8;
    const float* bp = bb + lane * 8;
#pragma unroll
    for (int i = 0; i < 8; ++i)
        dst[i] = (_Float16)((v[i] - mean) * rstd * gp[i] + bp[i]);
}

// ---------------------------------------------------------------------------
// Generic WMMA GEMM: Y[M][N] = X[M][K] @ Wt[N][K]^T + bias, fused epilogues.
// Block = 256 threads (8 waves), tile 128(M) x 64(N), BK=64, double-buffered
// async global->LDS staging overlapped with 8 WMMAs/wave per K-slice.
// EPI: 0 = QKV (bias, scale q cols, ->f16)
//      1 = PROJ (bias, window-reverse + roll(+3) + residual -> fp32 Yf)
//      2 = FC1  (bias, exact GELU, ->f16)
//      3 = FC2  (bias, Yf[idx] += v)
// ---------------------------------------------------------------------------
template <int EPI>
__global__ __launch_bounds__(256) void gemm_wmma_kernel(
    const _Float16* __restrict__ X, const _Float16* __restrict__ Wt,
    const float* __restrict__ bias,
    _Float16* __restrict__ Yh, float* __restrict__ Yf,
    const float* __restrict__ Xres, int N, int K) {

    __shared__ _Float16 As[2][128 * 64];   // 2 x 16 KB
    __shared__ _Float16 Bs[2][64 * 64];    // 2 x  8 KB

    const int t    = threadIdx.x;
    const int lane = t & 31;
    const int wave = t >> 5;
    const int wm   = wave >> 1;          // 0..3
    const int wn   = wave & 1;           // 0..1
    const int rowBase = blockIdx.y * 128;
    const int colBase = blockIdx.x * 64;
    const int lm   = lane & 15;
    const int kb   = (lane >> 4) * 8;    // A frag K sub-offset
    const int koff = (lane >> 4) * 16;   // B frag K offset
    const int hi8  = kb;                 // C/D M sub-offset

    // per-thread staging slices (6 async b128 ops per thread per K-slice)
    const int ar = t >> 1, ac = (t & 1) * 32;   // A: 128 rows, 32 halves/thr
    const int br = t >> 2, bc = (t & 3) * 16;   // B:  64 rows, 16 halves/thr
    const _Float16* agp = X  + (size_t)(rowBase + ar) * K + ac;
    const _Float16* bgp = Wt + (size_t)(colBase + br) * K + bc;

    v8f acc[2][2] = {};
    const int nK = K >> 6;

    // prologue: stage slice 0 into buffer 0
    {
#pragma unroll
        for (int c = 0; c < 4; ++c)
            async_cp_b128(agp + c * 8, &As[0][ar * 64 + ac + c * 8]);
#pragma unroll
        for (int c = 0; c < 2; ++c)
            async_cp_b128(bgp + c * 8, &Bs[0][br * 64 + bc + c * 8]);
    }

    for (int k0 = 0; k0 < nK; ++k0) {
        const int cur = k0 & 1;
        if (k0 + 1 < nK) {
            const int nxt = cur ^ 1;
            const int kk  = (k0 + 1) << 6;
#pragma unroll
            for (int c = 0; c < 4; ++c)
                async_cp_b128(agp + kk + c * 8, &As[nxt][ar * 64 + ac + c * 8]);
#pragma unroll
            for (int c = 0; c < 2; ++c)
                async_cp_b128(bgp + kk + c * 8, &Bs[nxt][br * 64 + bc + c * 8]);
            wait_async<6>();   // slice k0 landed; next slice may stay in flight
        } else {
            wait_async<0>();
        }
        __syncthreads();
#pragma unroll
        for (int kt = 0; kt < 2; ++kt) {
#pragma unroll
            for (int mt = 0; mt < 2; ++mt) {
                v16h af = ld_afrag(&As[cur][(wm * 32 + mt * 16 + lm) * 64 + kt * 32], kb);
#pragma unroll
                for (int nt = 0; nt < 2; ++nt) {
                    v16h bf = ld_bfrag(&Bs[cur][(wn * 32 + nt * 16 + lm) * 64 + kt * 32 + koff]);
                    acc[mt][nt] = __builtin_amdgcn_wmma_f32_16x16x32_f16(
                        false, af, false, bf, (short)0, acc[mt][nt], false, false);
                }
            }
        }
        __syncthreads();   // protect cur before it is refilled next iteration
    }

#pragma unroll
    for (int mt = 0; mt < 2; ++mt)
#pragma unroll
        for (int nt = 0; nt < 2; ++nt)
#pragma unroll
            for (int r = 0; r < 8; ++r) {
                const int row = rowBase + wm * 32 + mt * 16 + hi8 + r;
                const int col = colBase + wn * 32 + nt * 16 + lm;
                float v = acc[mt][nt][r] + bias[col];
                if (EPI == 0) {                      // QKV: scale q block
                    if (col < 256) v *= SCALE_Q;
                    Yh[(size_t)row * N + col] = (_Float16)v;
                } else if (EPI == 1) {               // PROJ: reverse + residual
                    int win = row / 49, nn = row % 49;
                    int b = win >> 6, widx = win & 63;
                    int hh = (widx >> 3) * 7 + nn / 7;
                    int ww = (widx & 7) * 7 + nn % 7;
                    int h = hh + 3; if (h >= 56) h -= 56;   // roll(+3)
                    int w = ww + 3; if (w >= 56) w -= 56;
                    size_t idx = ((size_t)b * 3136 + h * 56 + w) * CDIM + col;
                    Yf[idx] = Xres[idx] + v;
                } else if (EPI == 2) {               // FC1: exact GELU
                    float gsc = 0.5f * v * (1.0f + erff(v * 0.70710678118654752f));
                    Yh[(size_t)row * N + col] = (_Float16)gsc;
                } else {                             // FC2: residual into d_out
                    size_t idx = (size_t)row * N + col;
                    Yf[idx] = Yf[idx] + v;
                }
            }
}

// ---------------------------------------------------------------------------
// Windowed attention: one 128-thread block per (window, head). 1024 windows,
// 8 heads. q/k staged via async-to-LDS, v transposed via VALU; S = q k^T via
// WMMA; softmax (+analytic rel bias, +shift mask) row-per-thread; O = P V.
// ---------------------------------------------------------------------------
__global__ __launch_bounds__(128) void attn_kernel(
    const _Float16* __restrict__ qkv,   // (50176, 768) f16, q pre-scaled
    const float* __restrict__ mask,     // (64, 49, 49)
    const float* __restrict__ rel_bias, // (169, 8)
    _Float16* __restrict__ o) {         // (50176, 256) f16

    __shared__ _Float16 qs[64 * 32];
    __shared__ _Float16 ks[64 * 32];
    __shared__ _Float16 vt[32 * 64];    // v transposed: vt[d][n]
    __shared__ float    S[64 * 64];
    __shared__ _Float16 P[64 * 64];

    const int win  = blockIdx.x >> 3;
    const int head = blockIdx.x & 7;
    const int t    = threadIdx.x;
    const int lane = t & 31;
    const int wave = t >> 5;            // 0..3: 16-row M strip
    const int lm   = lane & 15;
    const int kb   = (lane >> 4) * 8;
    const int koff = (lane >> 4) * 16;

    // ---- stage q/k (async) and v (transposed, VALU) ----
    {
        const int n  = t >> 1;          // 0..63
        const int c0 = (t & 1) * 16;
        if (n < 49) {
            const _Float16* base = qkv + (size_t)(win * 49 + n) * 768 + head * 32 + c0;
            async_cp_b128(base,           qs + n * 32 + c0);
            async_cp_b128(base + 8,       qs + n * 32 + c0 + 8);
            async_cp_b128(base + 256,     ks + n * 32 + c0);
            async_cp_b128(base + 256 + 8, ks + n * 32 + c0 + 8);
#pragma unroll
            for (int i = 0; i < 16; ++i) vt[(c0 + i) * 64 + n] = base[512 + i];
        } else {
#pragma unroll
            for (int i = 0; i < 16; ++i) {
                qs[n * 32 + c0 + i] = (_Float16)0.f;
                ks[n * 32 + c0 + i] = (_Float16)0.f;
                vt[(c0 + i) * 64 + n] = (_Float16)0.f;
            }
        }
    }
    wait_async<0>();
    __syncthreads();

    // ---- S = q @ k^T (49x49 in 64x64 tiles, K=32) ----
    {
        v16h af = ld_afrag(qs + (wave * 16 + lm) * 32, kb);
#pragma unroll
        for (int nt = 0; nt < 4; ++nt) {
            v16h bf = ld_bfrag(ks + (nt * 16 + lm) * 32 + koff);
            v8f c = {};
            c = __builtin_amdgcn_wmma_f32_16x16x32_f16(false, af, false, bf,
                                                       (short)0, c, false, false);
#pragma unroll
            for (int r = 0; r < 8; ++r)
                S[(wave * 16 + kb + r) * 64 + nt * 16 + lm] = c[r];
        }
    }
    __syncthreads();

    // ---- softmax rows (+rel bias + shift mask), row per thread ----
    if (t < 64) {
        const int n = t;
        if (n < 49) {
            const int i = n / 7, j = n % 7;
            const float* mrow = mask + ((size_t)(win & 63) * 49 + n) * 49;
            float mx = -1e30f;
            for (int m = 0; m < 49; ++m) {
                int i2 = m / 7, j2 = m % 7;
                int rel = (i - i2 + 6) * 13 + (j - j2 + 6);
                float s = S[n * 64 + m] + rel_bias[rel * 8 + head] + mrow[m];
                S[n * 64 + m] = s;
                mx = fmaxf(mx, s);
            }
            float sum = 0.f;
            for (int m = 0; m < 49; ++m) {
                float e = __expf(S[n * 64 + m] - mx);
                S[n * 64 + m] = e;
                sum += e;
            }
            float rinv = 1.0f / sum;
            for (int m = 0; m < 49; ++m) P[n * 64 + m] = (_Float16)(S[n * 64 + m] * rinv);
            for (int m = 49; m < 64; ++m) P[n * 64 + m] = (_Float16)0.f;
        } else {
            for (int m = 0; m < 64; ++m) P[n * 64 + m] = (_Float16)0.f;
        }
    }
    __syncthreads();

    // ---- O = P @ V (M=49 pad 64, N=32, K=64 in two steps) ----
#pragma unroll
    for (int nt = 0; nt < 2; ++nt) {
        v8f c = {};
#pragma unroll
        for (int kt = 0; kt < 2; ++kt) {
            v16h af = ld_afrag(P + (wave * 16 + lm) * 64 + kt * 32, kb);
            v16h bf = ld_bfrag(vt + (nt * 16 + lm) * 64 + kt * 32 + koff);
            c = __builtin_amdgcn_wmma_f32_16x16x32_f16(false, af, false, bf,
                                                       (short)0, c, false, false);
        }
#pragma unroll
        for (int r = 0; r < 8; ++r) {
            int row = wave * 16 + kb + r;
            if (row < 49)
                o[(size_t)(win * 49 + row) * CDIM + head * 32 + nt * 16 + lm] = (_Float16)c[r];
        }
    }
}

// ---------------------------------------------------------------------------
// Launch
// ---------------------------------------------------------------------------
extern "C" void kernel_launch(void* const* d_in, const int* in_sizes, int n_in,
                              void* d_out, int out_size, void* d_ws, size_t ws_size,
                              hipStream_t stream) {
    const float* x       = (const float*)d_in[0];
    const float* mask    = (const float*)d_in[1];
    const float* norm1_g = (const float*)d_in[2];
    const float* norm1_b = (const float*)d_in[3];
    const float* qkv_w   = (const float*)d_in[4];
    const float* qkv_b   = (const float*)d_in[5];
    const float* rel_b   = (const float*)d_in[6];
    const float* proj_w  = (const float*)d_in[7];
    const float* proj_b  = (const float*)d_in[8];
    const float* norm2_g = (const float*)d_in[9];
    const float* norm2_b = (const float*)d_in[10];
    const float* fc1_w   = (const float*)d_in[11];
    const float* fc1_b   = (const float*)d_in[12];
    const float* fc2_w   = (const float*)d_in[13];
    const float* fc2_b   = (const float*)d_in[14];
    float* out = (float*)d_out;

    // ---- workspace carve (f16) ----
    char* w = (char*)d_ws;
    size_t off = 0;
    auto carve = [&](size_t nbytes) { char* p = w + off; off += (nbytes + 255) & ~(size_t)255; return p; };
    _Float16* wq   = (_Float16*)carve((size_t)768 * 256 * 2);
    _Float16* wp   = (_Float16*)carve((size_t)256 * 256 * 2);
    _Float16* w1   = (_Float16*)carve((size_t)1024 * 256 * 2);
    _Float16* w2   = (_Float16*)carve((size_t)256 * 1024 * 2);
    _Float16* hwin = (_Float16*)carve((size_t)MROWS * 256 * 2);
    _Float16* qkvh = (_Float16*)carve((size_t)MROWS * 768 * 2);
    _Float16* oh   = (_Float16*)carve((size_t)MROWS * 256 * 2);
    _Float16* h2   = (_Float16*)carve((size_t)MROWS * 256 * 2);
    _Float16* f1o  = hwin;   // FC1 out (M x 1024) overlays dead hwin+qkvh

    // 1) weights fp32 -> f16
    cvt_f16_kernel<<<(768 * 256 + 255) / 256, 256, 0, stream>>>(qkv_w, wq, 768 * 256);
    cvt_f16_kernel<<<(256 * 256 + 255) / 256, 256, 0, stream>>>(proj_w, wp, 256 * 256);
    cvt_f16_kernel<<<(1024 * 256 + 255) / 256, 256, 0, stream>>>(fc1_w, w1, 1024 * 256);
    cvt_f16_kernel<<<(256 * 1024 + 255) / 256, 256, 0, stream>>>(fc2_w, w2, 256 * 1024);

    // 2) LN1 + shift + window partition -> f16
    ln_kernel<<<MROWS / 8, 256, 0, stream>>>(x, norm1_g, norm1_b, hwin, 1);

    // 3) QKV GEMM (M=50176, K=256, N=768), bias + q-scale
    gemm_wmma_kernel<0><<<dim3(768 / 64, MROWS / 128), 256, 0, stream>>>(
        hwin, wq, qkv_b, qkvh, nullptr, nullptr, 768, 256);

    // 4) windowed attention (1024 windows x 8 heads)
    attn_kernel<<<1024 * 8, 128, 0, stream>>>(qkvh, mask, rel_b, oh);

    // 5) proj GEMM + window reverse + roll + residual -> d_out (= x2)
    gemm_wmma_kernel<1><<<dim3(256 / 64, MROWS / 128), 256, 0, stream>>>(
        oh, wp, proj_b, nullptr, out, x, 256, 256);

    // 6) LN2 on x2 -> f16
    ln_kernel<<<MROWS / 8, 256, 0, stream>>>(out, norm2_g, norm2_b, h2, 0);

    // 7) FC1 GEMM + GELU (M, K=256, N=1024)
    gemm_wmma_kernel<2><<<dim3(1024 / 64, MROWS / 128), 256, 0, stream>>>(
        h2, w1, fc1_b, f1o, nullptr, nullptr, 1024, 256);

    // 8) FC2 GEMM (M, K=1024, N=256), d_out += result
    gemm_wmma_kernel<3><<<dim3(256 / 64, MROWS / 128), 256, 0, stream>>>(
        f1o, w2, fc2_b, nullptr, out, nullptr, 256, 1024);
}
